// MultiHeadAttention_9242769622339
// MI455X (gfx1250) — matrix operand order repaired and checked by reading, not compiled, not run on hardware
//
#include <hip/hip_runtime.h>

// ---------------------------------------------------------------------------
// MI455X (gfx1250, wave32) multi-head causal attention, bf16 WMMA everywhere.
//   B=4, S=2048, D_IN=D_OUT=1024, H=16, Dh=64
// Pipeline:
//   1) qkv_gemm : Q/K/V = X @ W{q,k,v}   (fp32 -> bf16 staged, bf16 WMMA, bf16 out)
//   2) attn     : flash attention, score-transpose trick (S^T = K @ Q^T),
//                 K tile staged to LDS via the Tensor Data Mover (TDM) when the
//                 builtin is available (s_wait_tensorcnt sync), manual fallback.
//   3) oproj    : out = ctx @ Wo + bo    (fp32 out)
// ---------------------------------------------------------------------------

typedef __bf16 bf16;
typedef unsigned short u16;
typedef unsigned int u32;
typedef unsigned long long u64;
typedef __attribute__((ext_vector_type(4)))  u32   u32x4;
typedef __attribute__((ext_vector_type(4)))  u16   u16x4;
typedef __attribute__((ext_vector_type(4)))  float f32x4;
typedef __attribute__((ext_vector_type(16))) bf16  v16bf;
typedef __attribute__((ext_vector_type(8)))  float v8f;

// TDM descriptor argument vector types (per CDNA5_HIP.md probe results)
typedef __attribute__((ext_vector_type(4))) u32 tdm_v4u;
typedef __attribute__((ext_vector_type(8))) int tdm_v8i;
typedef __attribute__((ext_vector_type(4))) int tdm_v4i;

#if defined(__has_builtin)
#if __has_builtin(__builtin_amdgcn_tensor_load_to_lds) && \
    __has_builtin(__builtin_amdgcn_s_wait_tensorcnt)
#define USE_TDM 1
#endif
#endif
#ifndef USE_TDM
#define USE_TDM 0
#endif

#define DEVINL __device__ __forceinline__

static constexpr int Bc   = 4;
static constexpr int Sc   = 2048;
static constexpr int Dc   = 1024;   // D_IN == D_OUT
static constexpr int Hc   = 16;
static constexpr int Dh   = 64;
static constexpr int Mtot = Bc * Sc;  // 8192 tokens

struct b32bytes { u32x4 lo, hi; };

// Load a 32-byte WMMA fragment as two 16-byte chunks (handles the split
// K order when p1 = p0 + 16; p1 = p0 + 8 gives contiguous k).
DEVINL v16bf ld_frag(const u16* p0, const u16* p1) {
  b32bytes t;
  t.lo = *(const u32x4*)p0;
  t.hi = *(const u32x4*)p1;
  return __builtin_bit_cast(v16bf, t);
}

DEVINL u16 f2bf(float f) { return __builtin_bit_cast(u16, (bf16)f); }

DEVINL v8f wmma_bf16(v16bf a, v16bf b, v8f c) {
  // D = A(16x32 bf16) x B(32x16 bf16) + C(16x16 f32)
  return __builtin_amdgcn_wmma_f32_16x16x32_bf16(
      false, a, false, b, (short)0, c, false, false);
}

// ---------------------------------------------------------------------------
// Kernel 1: QKV projection.  C[z] = X(8192x1024 f32) @ W[z](1024x1024 f32),
// output bf16.  128x128 block tile, BK=32, 8 waves (2x4), wave tile 64x32.
// ---------------------------------------------------------------------------
__global__ __launch_bounds__(256) void qkv_gemm_kernel(
    const float* __restrict__ X,
    const float* __restrict__ W0, const float* __restrict__ W1,
    const float* __restrict__ W2,
    u16* __restrict__ O0, u16* __restrict__ O1, u16* __restrict__ O2)
{
  const float* W = (blockIdx.z == 0) ? W0 : (blockIdx.z == 1) ? W1 : W2;
  u16*         O = (blockIdx.z == 0) ? O0 : (blockIdx.z == 1) ? O1 : O2;

  const int n0   = blockIdx.x * 128;
  const int m0   = blockIdx.y * 128;
  const int t    = threadIdx.x;
  const int lane = t & 31;
  const int wv   = t >> 5;
  const int wm   = (wv >> 2) * 64;   // wave M offset: 0 / 64
  const int wn   = (wv & 3) * 32;    // wave N offset: 0,32,64,96
  const int lr   = lane & 15;
  const int hb   = lane >> 4;        // half-wave bit
  const int hs   = hb * 8;

  __shared__ __align__(16) u16 As[128 * 40];   // [m][k] bf16, stride 40
  __shared__ __align__(16) u16 Ws[128 * 40];   // [n][k] bf16 (transposed), stride 40

  v8f acc[4][2] = {};

  for (int kt = 0; kt < Dc / 32; ++kt) {
    // stage X tile 128x32: fp32 -> bf16
    #pragma unroll
    for (int i = 0; i < 4; ++i) {
      const int row = (t >> 3) + i * 32;
      const int c4  = (t & 7) * 4;
      f32x4 a = *(const f32x4*)(X + (size_t)(m0 + row) * Dc + kt * 32 + c4);
      u16x4 p;
      p[0] = f2bf(a[0]); p[1] = f2bf(a[1]); p[2] = f2bf(a[2]); p[3] = f2bf(a[3]);
      *(u16x4*)&As[row * 40 + c4] = p;
    }
    // stage W tile 32x128 transposed into [n][k]
    #pragma unroll
    for (int i = 0; i < 4; ++i) {
      const int k  = (t >> 5) + i * 8;
      const int c4 = (t & 31) * 4;
      f32x4 w = *(const f32x4*)(W + (size_t)(kt * 32 + k) * Dc + n0 + c4);
      #pragma unroll
      for (int j = 0; j < 4; ++j) Ws[(c4 + j) * 40 + k] = f2bf(w[j]);
    }
    // prefetch next k-tile (global_prefetch_b8)
    if (kt + 1 < Dc / 32) {
      __builtin_prefetch(X + (size_t)(m0 + (t >> 3)) * Dc + (kt + 1) * 32, 0, 1);
      __builtin_prefetch(W + (size_t)((kt + 1) * 32 + (t >> 5)) * Dc + n0, 0, 1);
    }
    __syncthreads();

    v16bf af[4];
    #pragma unroll
    for (int mf = 0; mf < 4; ++mf) {
      const int row = wm + mf * 16 + lr;
      af[mf] = ld_frag(&As[row * 40 + hs], &As[row * 40 + hs + 16]);
    }
    v16bf bfr[2];
    #pragma unroll
    for (int nf = 0; nf < 2; ++nf) {
      const int col = wn + nf * 16 + lr;
      bfr[nf] = ld_frag(&Ws[col * 40 + hs], &Ws[col * 40 + hs + 16]);
    }
    #pragma unroll
    for (int mf = 0; mf < 4; ++mf)
      #pragma unroll
      for (int nf = 0; nf < 2; ++nf)
        acc[mf][nf] = wmma_bf16(af[mf], bfr[nf], acc[mf][nf]);
    __syncthreads();
  }

  // epilogue: C layout lane holds N=lr, M = r + hb*8
  #pragma unroll
  for (int mf = 0; mf < 4; ++mf)
    #pragma unroll
    for (int nf = 0; nf < 2; ++nf)
      #pragma unroll
      for (int r = 0; r < 8; ++r) {
        const int row = m0 + wm + mf * 16 + r + hs;
        const int col = n0 + wn + nf * 16 + lr;
        O[(size_t)row * Dc + col] = f2bf(acc[mf][nf][r]);
      }
}

// ---------------------------------------------------------------------------
// Kernel 2: causal flash attention over bf16 Q/K/V ([B*S,1024], head = 64 cols).
// Score-transpose trick: S^T = K_tile @ Q^T so softmax stats are per-lane and
// exp'ed scores drop straight into the A-fragment layout for P @ V.
// Block = 128 queries (8 waves x 16 rows), key chunks of 32.
// K chunk staged by the Tensor Data Mover: 2D tile 64 elem x 32 rows, bf16,
// tensor_dim0_stride = 1024 elems, LDS padding 4 DWORDs every 32 DWORDs to
// reproduce the 144-byte padded Ks rows (stride 72 halfwords).
// ---------------------------------------------------------------------------
__global__ __launch_bounds__(256) void attn_kernel(
    const u16* __restrict__ Q, const u16* __restrict__ K,
    const u16* __restrict__ V, u16* __restrict__ Octx)
{
  const int h    = blockIdx.y;
  const int b    = blockIdx.z;
  const int qblk = blockIdx.x * 128;
  const int t    = threadIdx.x;
  const int lane = t & 31;
  const int wv   = t >> 5;
  const int lr   = lane & 15;
  const int hb   = lane >> 4;
  const int hs   = hb * 8;
  const int q0   = qblk + wv * 16;
  const size_t rowbase = (size_t)b * Sc;
  const int hcol = h * Dh;

  __shared__ __align__(16) u16 Ks[32 * 72];  // [key][d]  stride 72 (144B rows)
  __shared__ __align__(16) u16 Vt[64 * 40];  // [d][key]  stride 40

  // Q fragments (B-layout for K @ Q^T): lane = query column, contiguous d.
  v16bf qf[2];
  #pragma unroll
  for (int f = 0; f < 2; ++f) {
    const u16* qp = Q + (rowbase + q0 + lr) * Dc + hcol + f * 32 + hb * 16;
    qf[f] = ld_frag(qp, qp + 8);
  }

  v8f   o[4] = {};
  float m_prev = -1e30f, lsum = 0.f;

  const int cb_end = qblk + 128;  // causal: keys beyond block max query unneeded
  for (int cbase = 0; cbase < cb_end; cbase += 32) {
    {  // stage 32 keys of K (TDM or manual) and V (manual transpose)
      const int kl = t >> 3, c8 = t & 7;
      const size_t g = (rowbase + cbase + kl) * Dc + hcol + c8 * 8;
#if USE_TDM
      if (wv == 0) {
        // D# group 0: count=1, lds_addr, 57-bit global addr, type=2
        const u64 ga = (u64)(uintptr_t)(K + (rowbase + cbase) * Dc + hcol);
        const u32 lo = (u32)(uintptr_t)(void*)Ks;
        tdm_v4u g0 = {};
        g0[0] = 1u;                                   // count=1 (valid user D#)
        g0[1] = lo;                                   // lds_addr (bytes)
        g0[2] = (u32)ga;                              // global_addr[31:0]
        g0[3] = ((u32)(ga >> 32) & 0x01FFFFFFu) | (2u << 30);  // [56:32] | type=2
        // D# group 1: bf16 tile 64 x 32, row stride 1024 elems, LDS pad 16B/128B
        tdm_v8i g1 = {};
        g1[0] = (1 << 16)      // data_size = 1 -> 2 bytes
              | (1 << 20)      // pad_enable
              | (4 << 22)      // pad_interval: 32 DWORDs (=128B of data)
              | (3 << 25);     // pad_amount: 4 DWORDs (=16B) -> 144B rows
        g1[1] = (64 << 16);                        // tensor_dim0[15:0] = 64
        g1[2] = (Sc & 0xFFFF) << 16;               // tensor_dim1[15:0] = 2048
        g1[3] = (64 << 16);                        // tile_dim0 = 64
        g1[4] = 32;                                // tile_dim1 = 32 rows
        g1[5] = Dc;                                // tensor_dim0_stride = 1024
        tdm_v4i z4 = {};
#if __clang_major__ >= 23
        tdm_v8i z8 = {};
        __builtin_amdgcn_tensor_load_to_lds(g0, g1, z4, z4, z8, 0);
#else
        __builtin_amdgcn_tensor_load_to_lds(g0, g1, z4, z4, 0);
#endif
      }
#else
      u32x4 kvv = *(const u32x4*)(K + g);
      *(u32x4*)&Ks[kl * 72 + c8 * 8] = kvv;
#endif
      u32x4 vvv = *(const u32x4*)(V + g);
      union { u32x4 u; u16 s[8]; } vu; vu.u = vvv;
      #pragma unroll
      for (int j = 0; j < 8; ++j) Vt[(c8 * 8 + j) * 40 + kl] = vu.s[j];
#if USE_TDM
      if (wv == 0) __builtin_amdgcn_s_wait_tensorcnt(0);
#endif
    }
    __syncthreads();

    if (cbase <= q0 + 15) {
      // S^T accumulators: M = key (vgpr+half), N = query (lane)
      v8f s[2] = {};
      #pragma unroll
      for (int f = 0; f < 2; ++f)
        #pragma unroll
        for (int c = 0; c < 2; ++c) {
          const int row = c * 16 + lr;
          v16bf kf = ld_frag(&Ks[row * 72 + f * 32 + hb * 16],
                             &Ks[row * 72 + f * 32 + hb * 16 + 8]);
          s[c] = wmma_bf16(kf, qf[f], s[c]);
        }

      const int qg = q0 + lr;
      float mh = -1e30f;
      #pragma unroll
      for (int c = 0; c < 2; ++c)
        #pragma unroll
        for (int r = 0; r < 8; ++r) {
          const int kg = cbase + c * 16 + r + hs;
          float sv = s[c][r] * 0.125f;           // 1/sqrt(64)
          sv = (kg <= qg) ? sv : -1e30f;         // causal mask
          s[c][r] = sv;
          mh = fmaxf(mh, sv);
        }
      mh = fmaxf(mh, __shfl_xor(mh, 16, 32));    // combine key halves
      const float m_new = fmaxf(m_prev, mh);
      const float alpha = __expf(m_prev - m_new);

      float ssum = 0.f;
      v16bf pf;                                   // P in A-fragment layout
      #pragma unroll
      for (int c = 0; c < 2; ++c)
        #pragma unroll
        for (int r = 0; r < 8; ++r) {
          const float p = __expf(s[c][r] - m_new);
          ssum += p;
          pf[c * 8 + r] = (bf16)p;
        }
      ssum += __shfl_xor(ssum, 16, 32);
      lsum = lsum * alpha + ssum;
      m_prev = m_new;

      // rescale O (lane = d column, row q = r + hb*8 -> fetch that q's alpha)
      #pragma unroll
      for (int r = 0; r < 8; ++r) {
        const float aq = __shfl(alpha, r + hs, 16);
        #pragma unroll
        for (int cd = 0; cd < 4; ++cd) o[cd][r] *= aq;
      }
      // O += P @ V  (B-fragment from transposed V: keys hs..hs+7, hs+16..hs+23 per lane)
      #pragma unroll
      for (int cd = 0; cd < 4; ++cd) {
        const int d = cd * 16 + lr;
        v16bf vf = ld_frag(&Vt[d * 40 + hs], &Vt[d * 40 + hs + 16]);
        o[cd] = wmma_bf16(pf, vf, o[cd]);
      }
    }
    __syncthreads();
  }

  const float linv = 1.0f / lsum;
  #pragma unroll
  for (int r = 0; r < 8; ++r) {
    const float lq = __shfl(linv, r + hs, 16);
    const size_t row = (rowbase + q0 + r + hs) * Dc;
    #pragma unroll
    for (int cd = 0; cd < 4; ++cd)
      Octx[row + hcol + cd * 16 + lr] = f2bf(o[cd][r] * lq);
  }
}

// ---------------------------------------------------------------------------
// Kernel 3: output projection  out = ctx(bf16) @ Wo(f32->bf16) + bo, fp32 out.
// ---------------------------------------------------------------------------
__global__ __launch_bounds__(256) void oproj_gemm_kernel(
    const u16* __restrict__ Actx, const float* __restrict__ W,
    const float* __restrict__ bias, float* __restrict__ O)
{
  const int n0   = blockIdx.x * 128;
  const int m0   = blockIdx.y * 128;
  const int t    = threadIdx.x;
  const int lane = t & 31;
  const int wv   = t >> 5;
  const int wm   = (wv >> 2) * 64;
  const int wn   = (wv & 3) * 32;
  const int lr   = lane & 15;
  const int hb   = lane >> 4;
  const int hs   = hb * 8;

  __shared__ __align__(16) u16 As[128 * 40];
  __shared__ __align__(16) u16 Ws[128 * 40];

  v8f acc[4][2] = {};

  for (int kt = 0; kt < Dc / 32; ++kt) {
    #pragma unroll
    for (int i = 0; i < 4; ++i) {
      const int row = (t >> 3) + i * 32;
      const int c4  = (t & 7) * 4;
      u16x4 a = *(const u16x4*)(Actx + (size_t)(m0 + row) * Dc + kt * 32 + c4);
      *(u16x4*)&As[row * 40 + c4] = a;
    }
    #pragma unroll
    for (int i = 0; i < 4; ++i) {
      const int k  = (t >> 5) + i * 8;
      const int c4 = (t & 31) * 4;
      f32x4 w = *(const f32x4*)(W + (size_t)(kt * 32 + k) * Dc + n0 + c4);
      #pragma unroll
      for (int j = 0; j < 4; ++j) Ws[(c4 + j) * 40 + k] = f2bf(w[j]);
    }
    if (kt + 1 < Dc / 32) {
      __builtin_prefetch(Actx + (size_t)(m0 + (t >> 3)) * Dc + (kt + 1) * 32, 0, 1);
      __builtin_prefetch(W + (size_t)((kt + 1) * 32 + (t >> 5)) * Dc + n0, 0, 1);
    }
    __syncthreads();

    v16bf af[4];
    #pragma unroll
    for (int mf = 0; mf < 4; ++mf) {
      const int row = wm + mf * 16 + lr;
      af[mf] = ld_frag(&As[row * 40 + hs], &As[row * 40 + hs + 16]);
    }
    v16bf bfr[2];
    #pragma unroll
    for (int nf = 0; nf < 2; ++nf) {
      const int col = wn + nf * 16 + lr;
      bfr[nf] = ld_frag(&Ws[col * 40 + hs], &Ws[col * 40 + hs + 16]);
    }
    #pragma unroll
    for (int mf = 0; mf < 4; ++mf)
      #pragma unroll
      for (int nf = 0; nf < 2; ++nf)
        acc[mf][nf] = wmma_bf16(af[mf], bfr[nf], acc[mf][nf]);
    __syncthreads();
  }

  #pragma unroll
  for (int mf = 0; mf < 4; ++mf)
    #pragma unroll
    for (int nf = 0; nf < 2; ++nf)
      #pragma unroll
      for (int r = 0; r < 8; ++r) {
        const int row = m0 + wm + mf * 16 + r + hs;
        const int col = n0 + wn + nf * 16 + lr;
        O[(size_t)row * Dc + col] = acc[mf][nf][r] + bias[col];
      }
}

// ---------------------------------------------------------------------------
extern "C" void kernel_launch(void* const* d_in, const int* in_sizes, int n_in,
                              void* d_out, int out_size, void* d_ws, size_t ws_size,
                              hipStream_t stream) {
  (void)in_sizes; (void)n_in; (void)out_size; (void)ws_size;
  const float* x  = (const float*)d_in[0];
  const float* Wq = (const float*)d_in[1];
  const float* Wk = (const float*)d_in[2];
  const float* Wv = (const float*)d_in[3];
  const float* Wo = (const float*)d_in[4];
  const float* bo = (const float*)d_in[5];
  float*       out = (float*)d_out;

  // workspace: 4 bf16 buffers of [8192,1024] -> 64 MB total
  const size_t elems = (size_t)Mtot * Dc;
  u16* qb = (u16*)d_ws;
  u16* kb = qb + elems;
  u16* vb = kb + elems;
  u16* cb = vb + elems;

  dim3 gQKV(Dc / 128, Mtot / 128, 3);   // (8, 64, 3)
  qkv_gemm_kernel<<<gQKV, 256, 0, stream>>>(x, Wq, Wk, Wv, qb, kb, vb);

  dim3 gAtt(Sc / 128, Hc, Bc);          // (16, 16, 4)
  attn_kernel<<<gAtt, 256, 0, stream>>>(qb, kb, vb, cb);

  dim3 gO(Dc / 128, Mtot / 128, 1);     // (8, 64, 1)
  oproj_gemm_kernel<<<gO, 256, 0, stream>>>(cb, Wo, bo, out);
}
